// MultiHeadAttention_83339545411830
// MI455X (gfx1250) — matrix operand order repaired and checked by reading, not compiled, not run on hardware
//
#include <hip/hip_runtime.h>

#define DEV __device__ __forceinline__

typedef __attribute__((ext_vector_type(16))) __bf16 v16bf;
typedef __attribute__((ext_vector_type(8)))  float  v8f;

// Problem constants (fixed by the reference)
constexpr int BB = 2, NN = 2048, DD = 1024, HH = 16, HDim = 64;
constexpr int MT = BB * NN;                 // 4096 token rows
constexpr float SCALE = 0.125f;             // HD^-0.5 = 1/8 (exact power of two)

union Frag16 {
  uint4 q[2];
  v16bf v;
};

DEV unsigned short f2bf(float f) {
  union { float f; unsigned int u; } x{f};
  unsigned int r = (x.u + 0x7FFFu + ((x.u >> 16) & 1u)) >> 16;  // RNE
  return (unsigned short)r;
}

DEV v8f wmma_bf16(v16bf a, v16bf b, v8f c) {
  // D = A(16x32) * B(32x16) + C, f32 accumulate
  return __builtin_amdgcn_wmma_f32_16x16x32_bf16(
      /*neg_a=*/false, a, /*neg_b=*/false, b,
      /*c_mod=*/(short)0, c, /*reuse_a=*/false, /*reuse_b=*/false);
}

// A-matrix 16x32 bf16 fragment: lane holds row (lane&15); VGPR0-7 hold
// K = hi*16 + 0..15 contiguous  (hi = lane>>4).
DEV v16bf load_afrag(const unsigned short* rowp, int hi) {
  Frag16 f;
  f.q[0] = *(const uint4*)(rowp + hi * 16);
  f.q[1] = *(const uint4*)(rowp + hi * 16 + 8);
  return f.v;
}

// B-matrix 32x16 bf16 fragment: lane holds column (lane&15); K = hi*16 + 0..15
// contiguous. p points at &col_row[k_base + hi*16]; 16 contiguous halves.
DEV v16bf load_bfrag(const unsigned short* p) {
  Frag16 f;
  f.q[0] = *(const uint4*)(p);
  f.q[1] = *(const uint4*)(p + 8);
  return f.v;
}

DEV v16bf ones_bfrag() {
  Frag16 f;
  const unsigned int one2 = 0x3F803F80u;  // two bf16 1.0
  f.q[0] = make_uint4(one2, one2, one2, one2);
  f.q[1] = make_uint4(one2, one2, one2, one2);
  return f.v;
}

// Async global->LDS copy (ASYNCcnt-tracked, VGLOBAL "GV" addressing):
// 16 bytes per lane from gaddr into LDS byte offset lds_off.
DEV void async_ld16(unsigned lds_off, const void* gaddr) {
  asm volatile("global_load_async_to_lds_b128 %0, %1, off" ::"v"(lds_off),
               "v"(gaddr)
               : "memory");
}
DEV void wait_async0() { asm volatile("s_wait_asynccnt 0" ::: "memory"); }

// Max-reduction across a 16-lane half using DPP16 butterflies (pure VALU;
// co-executes with the XDL/WMMA pipe instead of burning the LDS pipe).
#if __has_builtin(__builtin_amdgcn_update_dpp)
DEV float half_max16(float x) {
  int v, t;
  v = __builtin_bit_cast(int, x);
  t = __builtin_amdgcn_update_dpp(0, v, 0xB1, 0xF, 0xF, true);   // quad_perm xor1
  x = fmaxf(x, __builtin_bit_cast(float, t));
  v = __builtin_bit_cast(int, x);
  t = __builtin_amdgcn_update_dpp(0, v, 0x4E, 0xF, 0xF, true);   // quad_perm xor2
  x = fmaxf(x, __builtin_bit_cast(float, t));
  v = __builtin_bit_cast(int, x);
  t = __builtin_amdgcn_update_dpp(0, v, 0x141, 0xF, 0xF, true);  // row_half_mirror
  x = fmaxf(x, __builtin_bit_cast(float, t));
  v = __builtin_bit_cast(int, x);
  t = __builtin_amdgcn_update_dpp(0, v, 0x140, 0xF, 0xF, true);  // row_mirror
  x = fmaxf(x, __builtin_bit_cast(float, t));
  return x;
}
#else
DEV float half_max16(float x) {
  for (int off = 1; off < 16; off <<= 1) x = fmaxf(x, __shfl_xor(x, off, 32));
  return x;
}
#endif

// ---------------------------------------------------------------------------
// fp32 -> bf16 conversion
// ---------------------------------------------------------------------------
__global__ void cvt_bf16(const float* __restrict__ src,
                         unsigned short* __restrict__ dst, int n) {
  int i = blockIdx.x * blockDim.x + threadIdx.x;
  int stride = gridDim.x * blockDim.x;
  for (; i < n; i += stride) dst[i] = f2bf(src[i]);
}

// ---------------------------------------------------------------------------
// GEMM: C[M][O] = (A(MxK) @ W(OxK)^T + bias) * oscale
// MODE 0: store bf16 Q/K layout  [B][H][N][HD]
// MODE 1: store bf16 V transposed [B][H][HD][N]
// MODE 2: store fp32 row-major   [M][O]  (final output)
// ---------------------------------------------------------------------------
constexpr int MODE_QK = 0, MODE_VT = 1, MODE_OUT = 2;

template <int MODE>
DEV void store_acc(void* dst, v8f c, float bo, float oscale, int mrow0,
                   int ocol, int hi) {
  // f32 C/D layout: VGPR j -> M = j + 8*hi, N = lane&15 (= ocol's low part)
#pragma unroll
  for (int j = 0; j < 8; ++j) {
    float v = (c[j] + bo) * oscale;
    int m = mrow0 + j + 8 * hi;
    if constexpr (MODE == MODE_OUT) {
      ((float*)dst)[(size_t)m * DD + ocol] = v;
    } else {
      int b = m >> 11, n = m & (NN - 1);          // N = 2048
      int h = ocol >> 6, hd = ocol & (HDim - 1);  // HD = 64
      if constexpr (MODE == MODE_QK) {
        ((unsigned short*)dst)[(((size_t)(b * HH + h) * NN + n) << 6) + hd] =
            f2bf(v);
      } else {
        ((unsigned short*)dst)[((size_t)(b * HH + h) * HDim + hd) * NN + n] =
            f2bf(v);
      }
    }
  }
}

template <int MODE>
__global__ __launch_bounds__(256) void gemm_qkv(
    const unsigned short* __restrict__ A,  // MT x DD bf16
    const unsigned short* __restrict__ W,  // DD x DD bf16 (row-major [o][k])
    const float* __restrict__ bias,        // DD
    void* __restrict__ dst, float oscale) {
  const int lane = threadIdx.x & 31;
  const int wave = threadIdx.x >> 5;
  const int hi = lane >> 4, ln = lane & 15;
  const int waveM = wave >> 1, waveO = wave & 1;  // 4x2 waves -> 128x64 tile
  const int m0 = blockIdx.x * 128 + waveM * 32;
  const int o0 = blockIdx.y * 64 + waveO * 32;

  const unsigned short* a0p = A + (size_t)(m0 + ln) * DD;
  const unsigned short* a1p = A + (size_t)(m0 + 16 + ln) * DD;
  const unsigned short* w0p = W + (size_t)(o0 + ln) * DD;
  const unsigned short* w1p = W + (size_t)(o0 + 16 + ln) * DD;

  v8f c00{}, c01{}, c10{}, c11{};
#pragma unroll 4
  for (int kk = 0; kk < DD; kk += 32) {
    __builtin_prefetch(a0p + kk + 256, 0, 1);  // global_prefetch_b8
    v16bf a0 = load_afrag(a0p + kk, hi);
    v16bf a1 = load_afrag(a1p + kk, hi);
    v16bf b0 = load_bfrag(w0p + kk + hi * 16);
    v16bf b1 = load_bfrag(w1p + kk + hi * 16);
    c00 = wmma_bf16(a0, b0, c00);
    c01 = wmma_bf16(a0, b1, c01);
    c10 = wmma_bf16(a1, b0, c10);
    c11 = wmma_bf16(a1, b1, c11);
  }
  float bo0 = bias[o0 + ln];
  float bo1 = bias[o0 + 16 + ln];
  store_acc<MODE>(dst, c00, bo0, oscale, m0, o0 + ln, hi);
  store_acc<MODE>(dst, c01, bo1, oscale, m0, o0 + 16 + ln, hi);
  store_acc<MODE>(dst, c10, bo0, oscale, m0 + 16, o0 + ln, hi);
  store_acc<MODE>(dst, c11, bo1, oscale, m0 + 16, o0 + 16 + ln, hi);
}

// ---------------------------------------------------------------------------
// Flash attention: per block = one (b,h), 128 query rows; per wave = 16 rows.
// K/V tiles are staged block-wide into LDS with double-buffered async copies
// (one global fetch per tile instead of one per wave).
// Q (pre-scaled by 1/8): [BH][N][HD], K: [BH][N][HD], Vt: [BH][HD][N],
// Y: [B*N][D] bf16 (heads merged).
// ---------------------------------------------------------------------------
__global__ __launch_bounds__(256) void attn_fmha(
    const unsigned short* __restrict__ Q, const unsigned short* __restrict__ K,
    const unsigned short* __restrict__ Vt, unsigned short* __restrict__ Y) {
  __shared__ unsigned short KtS[2][32 * HDim];  // [buf][key][hd]  2 x 4KB
  __shared__ unsigned short VtS[2][HDim * 32];  // [buf][hd][key]  2 x 4KB
  __shared__ unsigned short Pl[8][16 * 32];     // per-wave P tile     8KB

  const int tid = threadIdx.x;
  const int lane = tid & 31;
  const int wave = tid >> 5;
  const int hi = lane >> 4, ln = lane & 15;
  const int bh = blockIdx.y;
  const int b = bh >> 4, h = bh & 15;
  const int q0 = blockIdx.x * 128 + wave * 16;

  // Staging geometry: K tile 32x64 halves = 256 x 16B; V tile 64x32 = 256 x 16B
  const unsigned short* Kg = K + (size_t)bh * NN * HDim;
  const unsigned short* Vg = Vt + (size_t)bh * HDim * NN;
  const unsigned kls = (unsigned)(size_t)(&KtS[0][0]) + tid * 16;  // LDS bytes
  const unsigned vls = (unsigned)(size_t)(&VtS[0][0]) + tid * 16;
  const int krow = tid >> 3, kchunk = (tid & 7) * 8;  // K: 8 chunks per row
  const int vrow = tid >> 2, vchunk = (tid & 3) * 8;  // V: 4 chunks per row

  // Resident Q fragments: contraction HD=64 -> two 32-wide K-steps
  const unsigned short* Qp = Q + ((size_t)bh * NN + q0) * HDim;
  v16bf qa0 = load_afrag(Qp + ln * HDim, hi);
  v16bf qa1 = load_afrag(Qp + ln * HDim + 32, hi);
  const v16bf ones = ones_bfrag();

  v8f acc0{}, acc1{}, acc2{}, acc3{};  // O tiles: hd columns 0..63 in 16s
  float mi[8], li[8];
#pragma unroll
  for (int j = 0; j < 8; ++j) {
    mi[j] = -1e30f;
    li[j] = 0.0f;
  }

  unsigned short* Pw = &Pl[wave][0];

  // Prologue: stage tile 0 into buffer 0
  async_ld16(kls, Kg + (size_t)krow * HDim + kchunk);
  async_ld16(vls, Vg + (size_t)vrow * NN + vchunk);

  for (int kt = 0; kt < NN / 32; ++kt) {
    const int buf = kt & 1;
    wait_async0();
    __syncthreads();  // staged tile visible block-wide

    if (kt + 1 < NN / 32) {  // kick next tile into the other buffer
      const int ky2 = (kt + 1) * 32;
      async_ld16(kls + (buf ^ 1) * 4096,
                 Kg + ((size_t)ky2 + krow) * HDim + kchunk);
      async_ld16(vls + (buf ^ 1) * 4096,
                 Vg + (size_t)vrow * NN + ky2 + vchunk);
    }

    const unsigned short* Ks = &KtS[buf][0];
    const unsigned short* Vs = &VtS[buf][0];

    // ---- S = Q @ K^T for 32 keys (two 16-col tiles); scale pre-baked ----
    v16bf kb0a = load_bfrag(Ks + ln * HDim + hi * 16);
    v16bf kb0b = load_bfrag(Ks + ln * HDim + 32 + hi * 16);
    v16bf kb1a = load_bfrag(Ks + (16 + ln) * HDim + hi * 16);
    v16bf kb1b = load_bfrag(Ks + (16 + ln) * HDim + 32 + hi * 16);

    v8f s0{}, s1{};
    s0 = wmma_bf16(qa0, kb0a, s0);
    s0 = wmma_bf16(qa1, kb0b, s0);
    s1 = wmma_bf16(qa0, kb1a, s1);
    s1 = wmma_bf16(qa1, kb1b, s1);

    // ---- online softmax: max via DPP butterflies, sum via WMMA(P,ones) ----
    float corr[8];
#pragma unroll
    for (int j = 0; j < 8; ++j) {
      float x0 = s0[j], x1 = s1[j];
      float mx = half_max16(fmaxf(x0, x1));  // row lives in one 16-lane half
      float mn = fmaxf(mi[j], mx);
      corr[j] = __expf(mi[j] - mn);
      mi[j] = mn;
      float p0 = __expf(x0 - mn), p1 = __expf(x1 - mn);
      acc0[j] *= corr[j];
      acc1[j] *= corr[j];
      acc2[j] *= corr[j];
      acc3[j] *= corr[j];
      int m = j + 8 * hi;  // row index in D-layout
      Pw[m * 32 + ln] = f2bf(p0);
      Pw[m * 32 + 16 + ln] = f2bf(p1);
    }
    asm volatile("" ::: "memory");  // keep LDS store->load order (wave-private)

    // ---- reload P as an A-fragment (16 rows x 32 keys) from LDS ----
    Frag16 pf;
    {
      const unsigned short* pr = Pw + ln * 32;
      pf.q[0] = *(const uint4*)(pr + hi * 16);
      pf.q[1] = *(const uint4*)(pr + hi * 16 + 8);
    }
    asm volatile("" ::: "memory");

    // ---- O += P @ V; row-sums of P via one extra WMMA against ones ----
    v16bf vb0 = load_bfrag(Vs + (size_t)(ln)*32 + hi * 16);
    v16bf vb1 = load_bfrag(Vs + (size_t)(16 + ln) * 32 + hi * 16);
    v16bf vb2 = load_bfrag(Vs + (size_t)(32 + ln) * 32 + hi * 16);
    v16bf vb3 = load_bfrag(Vs + (size_t)(48 + ln) * 32 + hi * 16);
    acc0 = wmma_bf16(pf.v, vb0, acc0);
    acc1 = wmma_bf16(pf.v, vb1, acc1);
    acc2 = wmma_bf16(pf.v, vb2, acc2);
    acc3 = wmma_bf16(pf.v, vb3, acc3);
    v8f ls{};
    ls = wmma_bf16(pf.v, ones, ls);  // every column of D = row-sum of P
#pragma unroll
    for (int j = 0; j < 8; ++j) li[j] = li[j] * corr[j] + ls[j];

    __syncthreads();  // all reads done before next tile overwrites buffers
  }

  // ---- normalize and store Y[b*N+q][h*HD + hd] ----
#pragma unroll
  for (int j = 0; j < 8; ++j) {
    float inv = 1.0f / li[j];
    int gq = q0 + j + 8 * hi;
    size_t row = ((size_t)b * NN + gq) * DD + h * HDim + ln;
    Y[row] = f2bf(acc0[j] * inv);
    Y[row + 16] = f2bf(acc1[j] * inv);
    Y[row + 32] = f2bf(acc2[j] * inv);
    Y[row + 48] = f2bf(acc3[j] * inv);
  }
}

// ---------------------------------------------------------------------------
// Host-side orchestration
// ---------------------------------------------------------------------------
extern "C" void kernel_launch(void* const* d_in, const int* in_sizes, int n_in,
                              void* d_out, int out_size, void* d_ws,
                              size_t ws_size, hipStream_t stream) {
  const float* x = (const float*)d_in[0];
  const float* Wq = (const float*)d_in[1];
  const float* bq = (const float*)d_in[2];
  const float* Wk = (const float*)d_in[3];
  const float* bk = (const float*)d_in[4];
  const float* Wv = (const float*)d_in[5];
  const float* bv = (const float*)d_in[6];
  const float* Wo = (const float*)d_in[7];
  const float* bo = (const float*)d_in[8];

  unsigned short* ws = (unsigned short*)d_ws;
  const size_t XE = (size_t)MT * DD;  // 4M elems
  const size_t WE = (size_t)DD * DD;  // 1M elems
  unsigned short* xb = ws;            // x bf16
  unsigned short* Wqb = xb + XE;
  unsigned short* Wkb = Wqb + WE;
  unsigned short* Wvb = Wkb + WE;
  unsigned short* Wob = Wvb + WE;
  unsigned short* Qb = Wob + WE;   // [BH][N][HD], pre-scaled by 1/8
  unsigned short* Kb = Qb + XE;    // [BH][N][HD]
  unsigned short* Vtb = Kb + XE;   // [BH][HD][N]
  unsigned short* Yb = Vtb + XE;   // [B*N][D]

  cvt_bf16<<<1024, 256, 0, stream>>>(x, xb, (int)XE);
  cvt_bf16<<<512, 256, 0, stream>>>(Wq, Wqb, (int)WE);
  cvt_bf16<<<512, 256, 0, stream>>>(Wk, Wkb, (int)WE);
  cvt_bf16<<<512, 256, 0, stream>>>(Wv, Wvb, (int)WE);
  cvt_bf16<<<512, 256, 0, stream>>>(Wo, Wob, (int)WE);

  dim3 g(MT / 128, DD / 64);  // 32 x 16 blocks
  gemm_qkv<MODE_QK><<<g, 256, 0, stream>>>(xb, Wqb, bq, Qb, SCALE);
  gemm_qkv<MODE_QK><<<g, 256, 0, stream>>>(xb, Wkb, bk, Kb, 1.0f);
  gemm_qkv<MODE_VT><<<g, 256, 0, stream>>>(xb, Wvb, bv, Vtb, 1.0f);

  attn_fmha<<<dim3(NN / 128, BB * HH), 256, 0, stream>>>(Qb, Kb, Vtb, Yb);

  gemm_qkv<MODE_OUT><<<g, 256, 0, stream>>>(Yb, Wob, bo, d_out, 1.0f);
}
